// ScalarDotProductCriticNetworkRevised_78649441124615
// MI455X (gfx1250) — matrix operand order repaired and checked by reading, not compiled, not run on hardware
//
#include <hip/hip_runtime.h>
#include <hip/hip_bf16.h>

typedef _Float16 v16h __attribute__((ext_vector_type(16)));
typedef _Float16 v8h  __attribute__((ext_vector_type(8)));
typedef float    v8f  __attribute__((ext_vector_type(8)));

#define Bn   256
#define Nn   32
#define Dn   128
#define An   16
#define En   128
#define FDn  128
#define Hn   256
#define DOAP 160                       // DOA=144 padded to 160 (multiple of 32)
#define SCALE 0.08838834764831845f    // 1/sqrt(128)

// ---- workspace (f16, transposed weights), offsets in halves ----
#define WKP_OFF   0         // WkT_proc [128][128]
#define WQP_OFF   16384     // WqT_proc
#define WVP_OFF   32768     // WvT_proc
#define WKT_OFF   49152     // WkT
#define WQT_OFF   65536     // WqT
#define WET_OFF   81920     // WeT [128][160] (zero padded K>=144)
#define WAVT_OFF  102400    // WavT
#define W1T_OFF   118784    // W1T [256][128]

// ---- LDS arena (byte offsets), 64 KB, phase-overlaid ----
#define OFF_OAH  0        // [32][160] f16 oa ; later base_h [32][128] f16
#define OFF_OPH  10240    // [32][160] f16 op
#define OFF_Q1   20480    // [32][128] f16 q1h/q2h ; later av_oaT [128][32]
#define OFF_K1   28672    // [32][128] f16 k1h/k2h ; later diff_h [32][128]
#define OFF_V1T  36864    // [128][32] f16 v1T     ; later e_h [32][128]
#define OFF_S    45056    // [32][32] f32 logits
#define OFF_WH   49152    // [32][32] f16 softmax probs (w1 then w2)
#define OFF_WAV  51200    // [32][128] f16 wav1h ; later av_oa_h
#define OFF_BW1  10240    // [32][256] f16 baseW1h (overwrites OPH + head of Q1 slot)
#define OFF_DW1  36864    // [32][256] f16 diffW1h (overwrites V1T,S,WH,head of WAV)
#define OFF_W2S  59392    // [256] f32 W2
#define OFF_W2F  61440    // [32][32] f32 w2 (kept fp32 for final stage)

// output layout: value[256*1024] | w1[256*1024] | w2[256*1024]
#define OUT_W1 262144
#define OUT_W2 524288

// ---------------- WMMA tile core ----------------
// C_tile(16x16, f32) += A[m0..m0+15, :K] * Bt[n0..n0+15, :K]^T
// A row-major [M][K] (lds), Bt row-major [N][K] (lds or global f16).
__device__ __forceinline__ v8f mm_acc(const _Float16* A, int ldA,
                                      const _Float16* Bt, int ldB,
                                      int K, int m0, int n0, int lane) {
  const int r  = lane & 15;
  const int hi = lane >> 4;
  const _Float16* ap = A  + (m0 + r) * ldA + hi * 8;    // A: K split 0-7 / 16-23 per half-wave
  const _Float16* bp = Bt + (n0 + r) * ldB + hi * 8;    // B: K split 0-7 / 16-23 per half-wave
  v8f acc = {0.f, 0.f, 0.f, 0.f, 0.f, 0.f, 0.f, 0.f};
  for (int k = 0; k < K; k += 32) {
    v8h a0 = *(const v8h*)(ap + k);
    v8h a1 = *(const v8h*)(ap + k + 16);
    v8h b0 = *(const v8h*)(bp + k);
    v8h b1 = *(const v8h*)(bp + k + 16);
    v16h av = __builtin_shufflevector(a0, a1, 0,1,2,3,4,5,6,7,8,9,10,11,12,13,14,15);
    v16h bv = __builtin_shufflevector(b0, b1, 0,1,2,3,4,5,6,7,8,9,10,11,12,13,14,15);
    acc = __builtin_amdgcn_wmma_f32_16x16x32_f16(false, av, false, bv,
                                                 (short)0, acc, false, false);
  }
  return acc;
}

// M is always 32 (2 row-tiles). NT = N/16 column tiles. 4 waves split the tiles.
__device__ __forceinline__ void gemm_f16(const _Float16* A, int ldA, const _Float16* Bt, int ldB,
                                         int K, int NT, _Float16* C, int ldC, int lane, int wave) {
  for (int t = wave; t < 2 * NT; t += 4) {
    const int m0 = (t & 1) * 16, n0 = (t >> 1) * 16;
    v8f acc = mm_acc(A, ldA, Bt, ldB, K, m0, n0, lane);
    const int n  = n0 + (lane & 15);
    const int mb = m0 + ((lane >> 4) << 3);
#pragma unroll
    for (int v = 0; v < 8; ++v) C[(mb + v) * ldC + n] = (_Float16)acc[v];
  }
}

__device__ __forceinline__ void gemm_f16T(const _Float16* A, int ldA, const _Float16* Bt, int ldB,
                                          int K, int NT, _Float16* CT, int ldT, int lane, int wave) {
  for (int t = wave; t < 2 * NT; t += 4) {
    const int m0 = (t & 1) * 16, n0 = (t >> 1) * 16;
    v8f acc = mm_acc(A, ldA, Bt, ldB, K, m0, n0, lane);
    const int n  = n0 + (lane & 15);
    const int mb = m0 + ((lane >> 4) << 3);
#pragma unroll
    for (int v = 0; v < 8; ++v) CT[n * ldT + (mb + v)] = (_Float16)acc[v];
  }
}

__device__ __forceinline__ void gemm_f16NT(const _Float16* A, int ldA, const _Float16* Bt, int ldB,
                                           int K, int NT, _Float16* C, int ldC,
                                           _Float16* CT, int ldT, int lane, int wave) {
  for (int t = wave; t < 2 * NT; t += 4) {
    const int m0 = (t & 1) * 16, n0 = (t >> 1) * 16;
    v8f acc = mm_acc(A, ldA, Bt, ldB, K, m0, n0, lane);
    const int n  = n0 + (lane & 15);
    const int mb = m0 + ((lane >> 4) << 3);
#pragma unroll
    for (int v = 0; v < 8; ++v) {
      C[(mb + v) * ldC + n] = (_Float16)acc[v];
      CT[n * ldT + (mb + v)] = (_Float16)acc[v];
    }
  }
}

__device__ __forceinline__ void gemm_f32(const _Float16* A, int ldA, const _Float16* Bt, int ldB,
                                         int K, int NT, float* C, int ldC, int lane, int wave) {
  for (int t = wave; t < 2 * NT; t += 4) {
    const int m0 = (t & 1) * 16, n0 = (t >> 1) * 16;
    v8f acc = mm_acc(A, ldA, Bt, ldB, K, m0, n0, lane);
    const int n  = n0 + (lane & 15);
    const int mb = m0 + ((lane >> 4) << 3);
#pragma unroll
    for (int v = 0; v < 8; ++v) C[(mb + v) * ldC + n] = acc[v];
  }
}

// C = (f16)(A*Bt - sub)   (used for diff = av_op - av_oa)
__device__ __forceinline__ void gemm_diff(const _Float16* A, int ldA, const _Float16* Bt, int ldB,
                                          int K, int NT, const _Float16* sub, int ldS,
                                          _Float16* C, int ldC, int lane, int wave) {
  for (int t = wave; t < 2 * NT; t += 4) {
    const int m0 = (t & 1) * 16, n0 = (t >> 1) * 16;
    v8f acc = mm_acc(A, ldA, Bt, ldB, K, m0, n0, lane);
    const int n  = n0 + (lane & 15);
    const int mb = m0 + ((lane >> 4) << 3);
#pragma unroll
    for (int v = 0; v < 8; ++v)
      C[(mb + v) * ldC + n] = (_Float16)(acc[v] - (float)sub[(mb + v) * ldS + n]);
  }
}

// Row-wise softmax of s[32][32]*SCALE -> global out row, f16 probs, optional f32 copy.
__device__ __forceinline__ void softmax_rows(const float* s, _Float16* wh, float* wf,
                                             float* outg, int tid) {
  if (tid < Nn) {
    float row[Nn];
    float m = -1e30f;
#pragma unroll
    for (int j = 0; j < Nn; ++j) { float x = s[tid * Nn + j] * SCALE; row[j] = x; m = fmaxf(m, x); }
    float sum = 0.f;
#pragma unroll
    for (int j = 0; j < Nn; ++j) { float e = __expf(row[j] - m); row[j] = e; sum += e; }
    const float inv = 1.f / sum;
#pragma unroll
    for (int j = 0; j < Nn; ++j) {
      float p = row[j] * inv;
      outg[tid * Nn + j] = p;
      wh[tid * Nn + j] = (_Float16)p;
      if (wf) wf[tid * Nn + j] = p;
    }
  }
}

// ---------------- weight pre-pass: fp32 -> f16, transposed ----------------
// dst[e][k] = (k < R) ? src[k][e] : 0   ; dst is [C][ld]
__global__ void transpose_to_f16(const float* __restrict__ src, _Float16* __restrict__ dst,
                                 int R, int C, int ld) {
  int i = blockIdx.x * blockDim.x + threadIdx.x;
  if (i >= C * ld) return;
  int e = i / ld, k = i % ld;
  dst[i] = (_Float16)((k < R) ? src[k * C + e] : 0.0f);
}

// ---------------- main per-batch kernel ----------------
__global__ __launch_bounds__(128)
void critic_wmma_kernel(const float* __restrict__ states,
                        const float* __restrict__ policies,
                        const float* __restrict__ actions,
                        const float* __restrict__ W2,
                        const _Float16* __restrict__ wgt,
                        float* __restrict__ out) {
  __shared__ __align__(128) unsigned char smem[65536];
  _Float16* oah    = (_Float16*)(smem + OFF_OAH);
  _Float16* oph    = (_Float16*)(smem + OFF_OPH);
  _Float16* q1h    = (_Float16*)(smem + OFF_Q1);
  _Float16* k1h    = (_Float16*)(smem + OFF_K1);
  _Float16* v1T    = (_Float16*)(smem + OFF_V1T);
  float*    sf     = (float*)   (smem + OFF_S);
  _Float16* wh     = (_Float16*)(smem + OFF_WH);
  _Float16* wav1h  = (_Float16*)(smem + OFF_WAV);
  _Float16* q2h    = q1h;                              // reuse
  _Float16* k2h    = k1h;                              // reuse
  _Float16* eh     = v1T;                              // reuse
  _Float16* avoa   = wav1h;                            // reuse
  _Float16* avoaT  = q1h;                              // reuse
  _Float16* diffh  = k1h;                              // reuse
  _Float16* baseh  = oah;                              // reuse
  _Float16* bw1    = (_Float16*)(smem + OFF_BW1);
  _Float16* dw1    = (_Float16*)(smem + OFF_DW1);
  float*    W2s    = (float*)   (smem + OFF_W2S);
  float*    w2f    = (float*)   (smem + OFF_W2F);

  const int b    = blockIdx.x;
  const int tid  = threadIdx.x;
  const int lane = tid & 31;
  const int wave = tid >> 5;

  // --- fill oa/op (states|actions / states|policies, padded to 160) + W2 ---
  for (int i = tid; i < Nn * DOAP; i += 128) {
    const int r = i / DOAP, c = i % DOAP;
    float va, vp;
    if (c < Dn)            { float s = states[(b * Nn + r) * Dn + c]; va = s; vp = s; }
    else if (c < Dn + An)  { va = actions [(b * Nn + r) * An + (c - Dn)];
                             vp = policies[(b * Nn + r) * An + (c - Dn)]; }
    else                   { va = 0.f; vp = 0.f; }
    oah[i] = (_Float16)va;
    oph[i] = (_Float16)vp;
  }
  for (int i = tid; i < Hn; i += 128) W2s[i] = W2[i];
  __syncthreads();

  // --- stage 1: projections (A = oah cols 0..127) ---
  gemm_f16 (oah, DOAP, wgt + WQP_OFF, 128, 128, 8, q1h, 128, lane, wave);
  gemm_f16 (oah, DOAP, wgt + WKP_OFF, 128, 128, 8, k1h, 128, lane, wave);
  gemm_f16T(oah, DOAP, wgt + WVP_OFF, 128, 128, 8, v1T, 32,  lane, wave);   // store v1^T
  __syncthreads();

  gemm_f32(q1h, 128, k1h, 128, 128, 2, sf, Nn, lane, wave);                 // s1 = q1 k1^T
  __syncthreads();
  softmax_rows(sf, wh, nullptr, out + OUT_W1 + b * Nn * Nn, tid);           // w1
  __syncthreads();

  gemm_f16(wh, Nn, v1T, 32, Nn, 8, wav1h, 128, lane, wave);                 // wav1 = w1 v1
  __syncthreads();

  // --- stage 2 attention weights ---
  gemm_f16(wav1h, 128, wgt + WQT_OFF, 128, 128, 8, q2h, 128, lane, wave);
  gemm_f16(wav1h, 128, wgt + WKT_OFF, 128, 128, 8, k2h, 128, lane, wave);
  __syncthreads();
  gemm_f32(q2h, 128, k2h, 128, 128, 2, sf, Nn, lane, wave);                 // s2 = q2 k2^T
  __syncthreads();
  softmax_rows(sf, wh, w2f, out + OUT_W2 + b * Nn * Nn, tid);               // w2 (f16 + f32)
  __syncthreads();

  // --- factored av path: av_oa, diff = av_op - av_oa ---
  gemm_f16(oah, DOAP, wgt + WET_OFF, DOAP, DOAP, 8, eh, 128, lane, wave);   // e_oa = oa We
  __syncthreads();
  gemm_f16NT(eh, 128, wgt + WAVT_OFF, 128, 128, 8, avoa, 128, avoaT, 32, lane, wave);
  __syncthreads();
  gemm_f16(oph, DOAP, wgt + WET_OFF, DOAP, DOAP, 8, eh, 128, lane, wave);   // e_op = op We
  __syncthreads();
  gemm_diff(eh, 128, wgt + WAVT_OFF, 128, 128, 8, avoa, 128, diffh, 128, lane, wave);
  __syncthreads();
  gemm_f16(wh, Nn, avoaT, 32, Nn, 8, baseh, 128, lane, wave);               // base = w2 av_oa
  __syncthreads();

  // --- push W1 through the factorization ---
  gemm_f16(baseh, 128, wgt + W1T_OFF, 128, 128, 16, bw1, Hn, lane, wave);   // baseW1 [32][256]
  __syncthreads();
  gemm_f16(diffh, 128, wgt + W1T_OFF, 128, 128, 16, dw1, Hn, lane, wave);   // diffW1 [32][256]
  __syncthreads();

  // --- nonlinear stage: value[i,j] = sum_h leaky(baseW1[i,h] + w2[i,j]*diffW1[j,h]) * W2[h] ---
  for (int idx = tid; idx < Nn * Nn; idx += 128) {
    const int i = idx >> 5, j = idx & 31;
    const float wij = w2f[i * Nn + j];
    float acc = 0.f;
#pragma unroll 8
    for (int h = 0; h < Hn; ++h) {
      float t = fmaf(wij, (float)dw1[j * Hn + h], (float)bw1[i * Hn + h]);
      t = (t > 0.f) ? t : 0.01f * t;
      acc = fmaf(t, W2s[h], acc);
    }
    out[b * Nn * Nn + i * Nn + j] = acc;
  }
}

extern "C" void kernel_launch(void* const* d_in, const int* in_sizes, int n_in,
                              void* d_out, int out_size, void* d_ws, size_t ws_size,
                              hipStream_t stream) {
  const float* states   = (const float*)d_in[0];
  const float* policies = (const float*)d_in[1];
  const float* actions  = (const float*)d_in[2];
  _Float16* ws = (_Float16*)d_ws;

  struct { int src; int off; int R, C, ld; } cv[8] = {
    {3, WKP_OFF, 128, 128, 128}, {4, WQP_OFF, 128, 128, 128},
    {5, WVP_OFF, 128, 128, 128}, {6, WKT_OFF, 128, 128, 128},
    {7, WQT_OFF, 128, 128, 128}, {8, WET_OFF, 144, 128, 160},
    {9, WAVT_OFF, 128, 128, 128}, {10, W1T_OFF, 128, 256, 128},
  };
  for (int i = 0; i < 8; ++i) {
    const int total = cv[i].C * cv[i].ld;
    transpose_to_f16<<<(total + 255) / 256, 256, 0, stream>>>(
        (const float*)d_in[cv[i].src], ws + cv[i].off, cv[i].R, cv[i].C, cv[i].ld);
  }

  critic_wmma_kernel<<<Bn, 128, 0, stream>>>(states, policies, actions,
                                             (const float*)d_in[11], ws, (float*)d_out);
}
